// GNNModule_65790309040228
// MI455X (gfx1250) — matrix operand order repaired and checked by reading, not compiled, not run on hardware
//
#include <hip/hip_runtime.h>
#include <math.h>

typedef __attribute__((ext_vector_type(16))) _Float16 v16h;
typedef __attribute__((ext_vector_type(8)))  _Float16 v8h;
typedef __attribute__((ext_vector_type(8)))  float    v8f;

#define D 64
#define NGRAPH 64

// ---------------- utility kernels ----------------

__global__ void k_zero(float* p, int n) {
  int i = blockIdx.x * blockDim.x + threadIdx.x;
  if (i < n) p[i] = 0.0f;
}

__global__ void k_deg(const int* __restrict__ col, float* __restrict__ deg, int E) {
  int e = blockIdx.x * blockDim.x + threadIdx.x;
  if (e < E) atomicAdd(&deg[col[e]], 1.0f);
}

__global__ void k_dinv(float* __restrict__ deg, int N) {
  int i = blockIdx.x * blockDim.x + threadIdx.x;
  if (i < N) deg[i] = rsqrtf(deg[i] + 1.0f);   // deg incl. self-loop, >= 1
}

__global__ void k_cvt_f16(const float* __restrict__ src, _Float16* __restrict__ dst, int n) {
  int i = blockIdx.x * blockDim.x + threadIdx.x;
  if (i < n) dst[i] = (_Float16)src[i];
}

// Pack W [64][64] f32 row-major into WMMA B-fragment order:
// Wp[((ks*4+nt)*32 + lane)*16 + j] = (f16) W[ks*32 + hg*16 + j][nt*16 + nl]
// where hg = lane>>4, nl = lane&15.  8 fragments * 32 lanes * 16 halves = 8KB.
__global__ void k_pack_w(const float* __restrict__ W, _Float16* __restrict__ Wp) {
  int i = blockIdx.x * blockDim.x + threadIdx.x;   // 0..4095
  if (i >= 8 * 32 * 16) return;
  int fi   = i >> 9;         // fragment index: ks*4 + nt
  int lane = (i >> 4) & 31;
  int j    = i & 15;
  int ks = fi >> 2, nt = fi & 3;
  int hg = lane >> 4, nl = lane & 15;
  int ncol = nt * 16 + nl;
  int kk   = ks * 32 + hg * 16 + j;
  Wp[i] = (_Float16)W[kk * D + ncol];
}

// ---------------- WMMA GEMM + fused self-loop term ----------------
// t[N,64] = A[N,64] * W ;  agg[N,64] = t * dinv^2
// One wave per 16-row tile. K=64 -> 2 k-steps of 32. N=64 -> 4 accumulator tiles.
__global__ void k_gemm64(const _Float16* __restrict__ A, const _Float16* __restrict__ Wp,
                         const float* __restrict__ dinv,
                         float* __restrict__ Out, float* __restrict__ Agg, int N) {
  int wave   = (blockIdx.x * blockDim.x + threadIdx.x) >> 5;
  int ntiles = (N + 15) >> 4;
  if (wave >= ntiles) return;                 // wave-uniform: EXEC stays all-ones for WMMA

  int lane = threadIdx.x & 31;
  int nl   = lane & 15;                       // row within tile (A) / col within tile (B,D)
  int hg   = lane >> 4;                       // half-group (K striping)
  int rowbase = wave << 4;
  int nodeRow = rowbase + nl; if (nodeRow >= N) nodeRow = N - 1;
  const _Float16* ar = A + (size_t)nodeRow * D;

  union Frag { v16h v; v8h h[2]; };

  v8f acc[4];
  #pragma unroll
  for (int nt = 0; nt < 4; ++nt) acc[nt] = (v8f){0,0,0,0,0,0,0,0};

  #pragma unroll
  for (int ks = 0; ks < 2; ++ks) {
    // A fragment (contiguous K order, matching packed B): lane holds K = [ks*32+hg*16, +16)
    Frag a;
    a.h[0] = *(const v8h*)(ar + ks * 32 + hg * 16);
    a.h[1] = *(const v8h*)(ar + ks * 32 + hg * 16 + 8);
    #pragma unroll
    for (int nt = 0; nt < 4; ++nt) {
      const _Float16* bp = Wp + (size_t)((ks * 4 + nt) * 32 + lane) * 16;
      Frag b;
      b.h[0] = *(const v8h*)(bp);
      b.h[1] = *(const v8h*)(bp + 8);
      acc[nt] = __builtin_amdgcn_wmma_f32_16x16x32_f16(
          /*neg_a=*/false, a.v, /*neg_b=*/false, b.v,
          /*c_mod=*/(short)0, acc[nt], /*reuse_a=*/false, /*reuse_b=*/false);
    }
  }

  // Epilogue. D layout: VGPR r -> M = r + 8*hg, N = nl.
  // Full-tile fast path: single base pointer, all offsets compile-time constants.
  int mbase = rowbase + 8 * hg;
  if (rowbase + 16 <= N) {
    float*       tb = Out + (size_t)mbase * D + nl;
    float*       ab = Agg + (size_t)mbase * D + nl;
    const float* dv = dinv + mbase;
    float di2[8];
    #pragma unroll
    for (int r = 0; r < 8; ++r) { float d = dv[r]; di2[r] = d * d; }
    #pragma unroll
    for (int nt = 0; nt < 4; ++nt) {
      #pragma unroll
      for (int r = 0; r < 8; ++r) {
        float v = acc[nt][r];
        tb[(size_t)r * D + nt * 16] = v;
        ab[(size_t)r * D + nt * 16] = v * di2[r];
      }
    }
  } else {
    #pragma unroll
    for (int nt = 0; nt < 4; ++nt) {
      #pragma unroll
      for (int r = 0; r < 8; ++r) {
        int outRow = mbase + r;
        if (outRow < N) {
          float d = dinv[outRow];
          float v = acc[nt][r];
          Out[(size_t)outRow * D + nt * 16 + nl] = v;
          Agg[(size_t)outRow * D + nt * 16 + nl] = v * d * d;
        }
      }
    }
  }
}

// ---------------- aggregation kernels ----------------

// agg[col[e], f..f+3] += t[row[e], f..f+3] * dinv[row]*dinv[col]   (float4 per thread)
__global__ void k_scatter(const float* __restrict__ t, const float* __restrict__ dinv,
                          const int* __restrict__ row, const int* __restrict__ col,
                          float* __restrict__ agg, int E) {
  int id = blockIdx.x * blockDim.x + threadIdx.x;   // E*16 threads
  if (id >= E * 16) return;
  int e = id >> 4;
  int f = (id & 15) * 4;
  int r = row[e], c = col[e];
  float nrm = dinv[r] * dinv[c];
  const float4 v = *(const float4*)(t + (size_t)r * D + f);
  float* dst = agg + (size_t)c * D + f;
  atomicAdd(dst + 0, v.x * nrm);
  atomicAdd(dst + 1, v.y * nrm);
  atomicAdd(dst + 2, v.z * nrm);
  atomicAdd(dst + 3, v.w * nrm);
}

// x2h[i,f] = f16(relu(agg[i,f] + bias[f]))   (feeds next WMMA layer)
__global__ void k_relu_h16(const float* __restrict__ agg, const float* __restrict__ bias,
                           _Float16* __restrict__ outh, int ND) {
  int i = blockIdx.x * blockDim.x + threadIdx.x;
  if (i < ND) {
    float v = agg[i] + bias[i & (D - 1)];
    outh[i] = (_Float16)fmaxf(v, 0.0f);
  }
}

// h[i,f] = relu(agg[i,f] + bias[f])  (f32, feeds pooling)
__global__ void k_relu_f32(const float* __restrict__ agg, const float* __restrict__ bias,
                           float* __restrict__ outf, int ND) {
  int i = blockIdx.x * blockDim.x + threadIdx.x;
  if (i < ND) {
    float v = agg[i] + bias[i & (D - 1)];
    outf[i] = fmaxf(v, 0.0f);
  }
}

// ---------------- pooling ----------------

__global__ void k_pool(const float* __restrict__ h, const int* __restrict__ batch,
                       float* __restrict__ sums, float* __restrict__ cnt, int N) {
  int id = blockIdx.x * blockDim.x + threadIdx.x;   // N*16 threads
  if (id >= N * 16) return;
  int node = id >> 4;
  int f = (id & 15) * 4;
  int g = batch[node];
  const float4 v = *(const float4*)(h + (size_t)node * D + f);
  float* dst = sums + (size_t)g * D + f;
  atomicAdd(dst + 0, v.x);
  atomicAdd(dst + 1, v.y);
  atomicAdd(dst + 2, v.z);
  atomicAdd(dst + 3, v.w);
  if (f == 0) atomicAdd(&cnt[g], 1.0f);
}

__global__ void k_div(float* __restrict__ sums, const float* __restrict__ cnt, int n) {
  int i = blockIdx.x * blockDim.x + threadIdx.x;
  if (i < n) sums[i] = sums[i] / fmaxf(cnt[i / D], 1.0f);
}

// ---------------- launch ----------------

extern "C" void kernel_launch(void* const* d_in, const int* in_sizes, int n_in,
                              void* d_out, int out_size, void* d_ws, size_t ws_size,
                              hipStream_t stream) {
  (void)n_in; (void)ws_size;

  const float* x     = (const float*)d_in[0];
  const int*   ei    = (const int*)  d_in[1];   // edge_index [2,E] flat
  const int*   batch = (const int*)  d_in[2];
  const float* W1    = (const float*)d_in[3];
  const float* b1    = (const float*)d_in[4];
  const float* W2    = (const float*)d_in[5];
  const float* b2    = (const float*)d_in[6];

  const int N = in_sizes[0] / D;
  const int E = in_sizes[1] / 2;
  const int* row = ei;
  const int* col = ei + E;
  const int ND = N * D;

  // workspace carve-out (256B aligned)
  char* base = (char*)d_ws;
  size_t off = 0;
  auto carve = [&](size_t bytes) -> char* {
    char* p = base + off;
    off = (off + bytes + 255) & ~(size_t)255;
    return p;
  };
  float*    dinv = (float*)   carve((size_t)N * 4);          // deg -> dinv in place
  _Float16* xh   = (_Float16*)carve((size_t)ND * 2);         // f16 layer input (reused L2)
  _Float16* w1p  = (_Float16*)carve((size_t)8 * 32 * 16 * 2);// packed W1 fragments
  _Float16* w2p  = (_Float16*)carve((size_t)8 * 32 * 16 * 2);// packed W2 fragments
  float*    t    = (float*)   carve((size_t)ND * 4);         // GEMM out / final features
  float*    agg  = (float*)   carve((size_t)ND * 4);
  float*    cnt  = (float*)   carve((size_t)NGRAPH * 4);
  float*    out  = (float*)d_out;

  const int TB = 256;
  const int gN   = (N  + TB - 1) / TB;
  const int gE   = (E  + TB - 1) / TB;
  const int gND  = (ND + TB - 1) / TB;
  const int gE4  = (E * 16 + TB - 1) / TB;    // float4 scatter
  const int gN4  = (N * 16 + TB - 1) / TB;    // float4 pool
  const int ntiles = (N + 15) / 16;
  const int gGemm  = (ntiles + 3) / 4;        // 4 waves (128 threads) per block

  // degrees -> dinv
  k_zero<<<gN, TB, 0, stream>>>(dinv, N);
  k_deg <<<gE, TB, 0, stream>>>(col, dinv, E);
  k_dinv<<<gN, TB, 0, stream>>>(dinv, N);

  // conversions / weight packing
  k_cvt_f16<<<gND, TB, 0, stream>>>(x, xh, ND);
  k_pack_w <<<(8 * 32 * 16 + TB - 1) / TB, TB, 0, stream>>>(W1, w1p);
  k_pack_w <<<(8 * 32 * 16 + TB - 1) / TB, TB, 0, stream>>>(W2, w2p);

  // ---- layer 1 ----  (gemm writes t and agg = t*dinv^2 in one pass)
  k_gemm64  <<<gGemm, 128, 0, stream>>>(xh, w1p, dinv, t, agg, N);
  k_scatter <<<gE4, TB, 0, stream>>>(t, dinv, row, col, agg, E);
  k_relu_h16<<<gND, TB, 0, stream>>>(agg, b1, xh, ND);   // becomes layer-2 input

  // ---- layer 2 ----
  k_gemm64  <<<gGemm, 128, 0, stream>>>(xh, w2p, dinv, t, agg, N);
  k_scatter <<<gE4, TB, 0, stream>>>(t, dinv, row, col, agg, E);
  k_relu_f32<<<gND, TB, 0, stream>>>(agg, b2, t, ND);    // final node features (f32)

  // ---- global mean pool ----
  k_zero<<<(out_size + TB - 1) / TB, TB, 0, stream>>>(out, out_size);
  k_zero<<<1, NGRAPH, 0, stream>>>(cnt, NGRAPH);
  k_pool<<<gN4, TB, 0, stream>>>(t, batch, out, cnt, N);
  k_div <<<(out_size + TB - 1) / TB, TB, 0, stream>>>(out, cnt, out_size);
}
